// GEX_PPI_GAT_cat4_MLP_21655225106966
// MI455X (gfx1250) — compile-verified, matrix operand order checked by reading, not run on hardware
//
#include <hip/hip_runtime.h>
#include <hip/hip_bf16.h>

// ---------------- model dims ----------------
#define B_      32
#define G_      978
#define NN      (B_ * G_)       // 31296 batched nodes
#define NBITS_  2048
#define DRUG_   300
#define DG2V_   200
#define HEADS_  4
#define HID_    64
#define HD_     (HEADS_ * HID_) // 256
#define D1_     1465
#define D2_     882
#define R1_     652
#define R2_     326
#define PRED_   814             // DRUG + 2*HD + 2
#define P1_     542
#define P2_     271
#define NCLS_   2
#define CH_     (2 * HD_)       // 512 readout channels

// padded dims (multiples of 32)
#define PAD32(x) (((x) + 31) & ~31)
#define D1P   PAD32(D1_)    // 1472
#define D2P   PAD32(D2_)    // 896
#define DRUGP PAD32(DRUG_)  // 320
#define DG2VP PAD32(DG2V_)  // 224
#define GP    PAD32(G_)     // 992
#define R1P   PAD32(R1_)    // 672
#define R2P   PAD32(R2_)    // 352
#define PREDP PAD32(PRED_)  // 832
#define P1P   PAD32(P1_)    // 544
#define P2P   PAD32(P2_)    // 288
#define NCLSP 32
#define BP    256                        // padded batch rows (block tile)
#define NNP   (((NN + 255) / 256) * 256) // 31488 padded node rows

typedef __attribute__((ext_vector_type(16))) _Float16 v16h;
typedef __attribute__((ext_vector_type(8)))  _Float16 v8h;
typedef __attribute__((ext_vector_type(8)))  float    v8f;

__device__ __forceinline__ int imin(int a, int b) { return a < b ? a : b; }

// ---------------------------------------------------------------------------
// Guard-free GEMM: C[.,Np] = act(A[.,Kp] @ WtT + bias). All dims padded to
// tile multiples; weights pre-transposed/padded/zero-filled f16 Wt[Np][Kp].
// Block = 4 waves, tile 256(M) x 32(N); each wave = 4x2 16x16 WMMA tiles
// (8 x v_wmma_f32_16x16x32_f16 per 32-K step, full operand reuse).
// A-fragment = two aligned global b128 loads; B tile double-buffered in LDS
// (1 global b128 + 1 ds_store_b128 per thread, one barrier per K-step);
// b-fragment = two ds_load_b128.
// ---------------------------------------------------------------------------
#define GWAVES 4
#define GMT 4
#define GNT 2
#define GBM (GWAVES * GMT * 16)   // 256 rows / block
#define GBN (GNT * 16)            // 32 cols / block

template <typename TC>
__global__ void gemm16_wmma(const _Float16* __restrict__ A, int lda,
                            const _Float16* __restrict__ Wt, int Kp,
                            const float* __restrict__ bias, int N,
                            TC* __restrict__ C, int ldc, int relu)
{
    __shared__ _Float16 sBt[2][GBN * 32];   // ping-pong [n][k], 2x2KB
    const int lane = threadIdx.x & 31;
    const int wave = threadIdx.x >> 5;
    const int hh = lane >> 4;               // lane half
    const int ml = lane & 15;
    const int tn0 = blockIdx.x * GBN;
    const int tm0 = blockIdx.y * GBM + wave * (GMT * 16);

    // B staging: this thread owns column sn, kk in [skb, skb+8)
    const int sn  = threadIdx.x & 31;
    const int skb = (threadIdx.x >> 5) * 8;
    const _Float16* wsrc = Wt + (size_t)(tn0 + sn) * Kp + skb;

    const _Float16* arow[GMT];
#pragma unroll
    for (int mt = 0; mt < GMT; ++mt)
        arow[mt] = A + (size_t)(tm0 + mt * 16 + ml) * lda + hh * 8;

    v8f acc[GMT][GNT] = {};

    // prologue: stage first B tile
    *(v8h*)&sBt[0][sn * 32 + skb] = *(const v8h*)wsrc;
    __syncthreads();

    int cur = 0;
    for (int k0 = 0; k0 < Kp; k0 += 32) {
        // ---- b fragments from current buffer: two ds_load_b128 each ----
        v16h b[GNT];
#pragma unroll
        for (int nt = 0; nt < GNT; ++nt) {
            const v8h* bp = (const v8h*)&sBt[cur][(nt * 16 + ml) * 32 + hh * 8];
            v8h lo = bp[0];
            v8h hi = bp[2];
            b[nt] = __builtin_shufflevector(lo, hi, 0, 1, 2, 3, 4, 5, 6, 7,
                                            8, 9, 10, 11, 12, 13, 14, 15);
        }

        // ---- a fragments: two aligned global b128 loads each ----
        v16h a[GMT];
#pragma unroll
        for (int mt = 0; mt < GMT; ++mt) {
            const v8h* ap = (const v8h*)(arow[mt] + k0);
            v8h lo = ap[0];
            v8h hi = ap[2];
            a[mt] = __builtin_shufflevector(lo, hi, 0, 1, 2, 3, 4, 5, 6, 7,
                                            8, 9, 10, 11, 12, 13, 14, 15);
        }

        // ---- stage next B tile into the other buffer ----
        if (k0 + 32 < Kp)
            *(v8h*)&sBt[cur ^ 1][sn * 32 + skb] = *(const v8h*)(wsrc + k0 + 32);

        // ---- 8 WMMAs ----
#pragma unroll
        for (int mt = 0; mt < GMT; ++mt)
#pragma unroll
            for (int nt = 0; nt < GNT; ++nt)
                acc[mt][nt] = __builtin_amdgcn_wmma_f32_16x16x32_f16(
                    false, a[mt], false, b[nt], (short)0, acc[mt][nt],
                    false, false);

        __syncthreads();
        cur ^= 1;
    }

    // ---- store (C/D layout: VGPR r -> m = r + 8*half, n = lane&15) ----
#pragma unroll
    for (int nt = 0; nt < GNT; ++nt) {
        const int gn = tn0 + nt * 16 + ml;
        const float bv = bias ? bias[imin(gn, N - 1)] : 0.0f;
#pragma unroll
        for (int mt = 0; mt < GMT; ++mt)
#pragma unroll
            for (int r = 0; r < 8; ++r) {
                const int gm = tm0 + mt * 16 + r + hh * 8;
                float v = acc[mt][nt][r] + bv;
                if (relu) v = fmaxf(v, 0.0f);
                C[(size_t)gm * ldc + gn] = (TC)v;
            }
    }
}

// W[K,N] f32 -> Wt[Np,Kp] f16, transposed + zero-padded
__global__ void wconvT(const float* __restrict__ W, _Float16* __restrict__ Wt,
                       int K, int N, int Kp, int Np)
{
    long i = (long)blockIdx.x * blockDim.x + threadIdx.x;
    if (i >= (long)Np * Kp) return;
    int k = (int)(i % Kp);
    int n = (int)(i / Kp);
    float v = (k < K && n < N) ? W[(size_t)k * N + n] : 0.0f;
    Wt[i] = (_Float16)v;
}

// Training-mode BatchNorm1d over batch dim (biased variance), f16 output
__global__ void bn_kernel(const float* __restrict__ x,
                          const float* __restrict__ g,
                          const float* __restrict__ b,
                          _Float16* __restrict__ y, int rows, int cols)
{
    int c = blockIdx.x * blockDim.x + threadIdx.x;
    if (c >= cols) return;
    float m = 0.f;
    for (int r = 0; r < rows; ++r) m += x[(size_t)r * cols + c];
    m /= (float)rows;
    float v = 0.f;
    for (int r = 0; r < rows; ++r) {
        float d = x[(size_t)r * cols + c] - m;
        v += d * d;
    }
    v /= (float)rows;
    float inv = rsqrtf(v + 1e-5f) * g[c];
    for (int r = 0; r < rows; ++r)
        y[(size_t)r * cols + c] = (_Float16)((x[(size_t)r * cols + c] - m) * inv + b[c]);
}

// xf[n*256 + d] = gex_n[b,g] * g2v[g,d] (d<200), 0 for d in [200,224)
__global__ void gene_feat(const _Float16* __restrict__ gexn,
                          const float* __restrict__ g2v,
                          _Float16* __restrict__ xf)
{
    long i = (long)blockIdx.x * blockDim.x + threadIdx.x;
    if (i >= (long)NN * DG2VP) return;
    int d = (int)(i % DG2VP);
    long n = i / DG2VP;
    int g = (int)(n % G_);
    int b = (int)(n / G_);
    float v = (d < DG2V_)
                  ? (float)gexn[(size_t)b * G_ + g] * g2v[(size_t)g * DG2V_ + d]
                  : 0.0f;
    xf[n * HD_ + d] = (_Float16)v;
}

// per-(node, head) attention logits: als = <h, a_src>, ald = <h, a_dst>
__global__ void gat_alpha(const _Float16* __restrict__ h,
                          const float* __restrict__ a_src,
                          const float* __restrict__ a_dst,
                          float* __restrict__ als, float* __restrict__ ald)
{
    int i = blockIdx.x * blockDim.x + threadIdx.x;   // n*HEADS + head
    if (i >= NN * HEADS_) return;
    int hd = i & (HEADS_ - 1);
    int n  = i >> 2;
    const _Float16* hp = h + (size_t)n * HD_ + hd * HID_;
    const float* as = a_src + hd * HID_;
    const float* ad = a_dst + hd * HID_;
    float s = 0.f, d = 0.f;
    for (int k = 0; k < HID_; ++k) {
        float hv = (float)hp[k];
        s += hv * as[k];
        d += hv * ad[k];
    }
    als[i] = s; ald[i] = d;
}

// order-preserving float<->uint key for atomic max
__device__ __forceinline__ unsigned fkey(float f) {
    unsigned u = __float_as_uint(f);
    return (u & 0x80000000u) ? ~u : (u | 0x80000000u);
}
__device__ __forceinline__ float funkey(unsigned u) {
    return (u & 0x80000000u) ? __uint_as_float(u & 0x7fffffffu)
                             : __uint_as_float(~u);
}

__global__ void edge_logits(const int* __restrict__ src, const int* __restrict__ dst,
                            const float* __restrict__ als, const float* __restrict__ ald,
                            float* __restrict__ ev, unsigned* __restrict__ mkey, long E)
{
    long i = (long)blockIdx.x * blockDim.x + threadIdx.x;  // e*HEADS + head
    if (i >= E * HEADS_) return;
    int hd = (int)(i & (HEADS_ - 1));
    long e = i >> 2;
    int s = src[e], d = dst[e];
    float x = als[(size_t)s * HEADS_ + hd] + ald[(size_t)d * HEADS_ + hd];
    x = (x > 0.f) ? x : 0.2f * x;                           // LeakyReLU(0.2)
    ev[i] = x;
    atomicMax(&mkey[(size_t)d * HEADS_ + hd], fkey(x));
}

__global__ void edge_exp(const int* __restrict__ dst,
                         float* __restrict__ ev,
                         const unsigned* __restrict__ mkey,
                         float* __restrict__ z, long E)
{
    long i = (long)blockIdx.x * blockDim.x + threadIdx.x;
    if (i >= E * HEADS_) return;
    int hd = (int)(i & (HEADS_ - 1));
    long e = i >> 2;
    int d = dst[e];
    float m = funkey(mkey[(size_t)d * HEADS_ + hd]);
    float ex = __expf(ev[i] - m);
    ev[i] = ex;
    atomicAdd(&z[(size_t)d * HEADS_ + hd], ex);
}

__global__ void edge_agg(const int* __restrict__ src, const int* __restrict__ dst,
                         const float* __restrict__ ev, const float* __restrict__ z,
                         const _Float16* __restrict__ h, float* __restrict__ out, long E)
{
    long i = (long)blockIdx.x * blockDim.x + threadIdx.x;  // e*HEADS + head
    if (i >= E * HEADS_) return;
    int hd = (int)(i & (HEADS_ - 1));
    long e = i >> 2;
    int s = src[e], d = dst[e];
    float alpha = ev[i] / (z[(size_t)d * HEADS_ + hd] + 1e-16f);
    const _Float16* hs = h + (size_t)s * HD_ + hd * HID_;
    float* op = out + (size_t)d * HD_ + hd * HID_;
    for (int k = 0; k < HID_; ++k) atomicAdd(&op[k], alpha * (float)hs[k]);
}

// bias + ReLU; write next-hop features (f16, stride 256) + transposed rT rows
__global__ void gat_finish(const float* __restrict__ agg,
                           const float* __restrict__ bias,
                           _Float16* __restrict__ xout,
                           _Float16* __restrict__ rT, int hop)
{
    long i = (long)blockIdx.x * blockDim.x + threadIdx.x;  // n*HD + c
    if (i >= (long)NN * HD_) return;
    int c = (int)(i & (HD_ - 1));
    long n = i >> 8;
    float v = fmaxf(agg[i] + bias[c], 0.0f);
    xout[i] = (_Float16)v;
    int g = (int)(n % G_);
    int b = (int)(n / G_);
    rT[((size_t)b * CH_ + hop * HD_ + c) * GP + g] = (_Float16)v;
}

// readout layer 3: per-row dot with ReLU (N==1 output)
__global__ void ro3_dot(const _Float16* __restrict__ A, const float* __restrict__ w,
                        const float* __restrict__ b, _Float16* __restrict__ out,
                        int rows, int K, int lda)
{
    int r = blockIdx.x * blockDim.x + threadIdx.x;
    if (r >= rows) return;
    const _Float16* ap = A + (size_t)r * lda;
    float s = b[0];
    for (int k = 0; k < K; ++k) s += (float)ap[k] * w[k];
    out[r] = (_Float16)fmaxf(s, 0.0f);
}

__global__ void concat_t(const _Float16* __restrict__ demb, const _Float16* __restrict__ rout,
                         const _Float16* __restrict__ dosen, const _Float16* __restrict__ durn,
                         _Float16* __restrict__ t)
{
    int i = blockIdx.x * blockDim.x + threadIdx.x;
    if (i >= B_ * PRED_) return;
    int c = i % PRED_;
    int b = i / PRED_;
    _Float16 v;
    if (c < DRUG_)              v = demb[(size_t)b * DRUGP + c];
    else if (c < DRUG_ + CH_)   v = rout[(size_t)b * CH_ + (c - DRUG_)];
    else if (c == DRUG_ + CH_)  v = dosen[b];
    else                        v = durn[b];
    t[(size_t)b * PREDP + c] = v;
}

// copy [32,2] out of padded [256,32] f32 tile
__global__ void copy_out(const float* __restrict__ T3, float* __restrict__ out)
{
    int i = threadIdx.x;
    if (i >= B_ * NCLS_) return;
    int b = i / NCLS_, c = i % NCLS_;
    out[i] = T3[(size_t)b * NCLSP + c];
}

// ---------------------------------------------------------------------------
static inline void launch_gemm_h(const _Float16* A, int lda, const _Float16* Wt, int Kp,
                                 const float* bias, int N, _Float16* C, int ldc,
                                 int Mp, int Np, int relu, hipStream_t s)
{
    dim3 grid(Np / GBN, Mp / GBM);
    gemm16_wmma<_Float16><<<grid, GWAVES * 32, 0, s>>>(A, lda, Wt, Kp, bias, N, C, ldc, relu);
}
static inline void launch_gemm_f(const _Float16* A, int lda, const _Float16* Wt, int Kp,
                                 const float* bias, int N, float* C, int ldc,
                                 int Mp, int Np, int relu, hipStream_t s)
{
    dim3 grid(Np / GBN, Mp / GBM);
    gemm16_wmma<float><<<grid, GWAVES * 32, 0, s>>>(A, lda, Wt, Kp, bias, N, C, ldc, relu);
}

extern "C" void kernel_launch(void* const* d_in, const int* in_sizes, int n_in,
                              void* d_out, int out_size, void* d_ws, size_t ws_size,
                              hipStream_t stream)
{
    (void)n_in; (void)out_size; (void)ws_size;
    const float* drug_in = (const float*)d_in[0];
    const float* gex_in  = (const float*)d_in[1];
    const float* dose    = (const float*)d_in[2];
    const float* dur     = (const float*)d_in[3];
    const int*   ei      = (const int*)d_in[4];
    const float* bn1_g = (const float*)d_in[5],  *bn1_b = (const float*)d_in[6];
    const float* bn2_g = (const float*)d_in[7],  *bn2_b = (const float*)d_in[8];
    const float* bn3_g = (const float*)d_in[9],  *bn3_b = (const float*)d_in[10];
    const float* bn4_g = (const float*)d_in[11], *bn4_b = (const float*)d_in[12];
    const float* dm1_w = (const float*)d_in[13], *dm1_b = (const float*)d_in[14];
    const float* dm2_w = (const float*)d_in[15], *dm2_b = (const float*)d_in[16];
    const float* dm3_w = (const float*)d_in[17], *dm3_b = (const float*)d_in[18];
    const float* g2v   = (const float*)d_in[19];
    const float* gat0_w = (const float*)d_in[20], *gat0_as = (const float*)d_in[21];
    const float* gat0_ad = (const float*)d_in[22], *gat0_b = (const float*)d_in[23];
    const float* gat1_w = (const float*)d_in[24], *gat1_as = (const float*)d_in[25];
    const float* gat1_ad = (const float*)d_in[26], *gat1_b = (const float*)d_in[27];
    const float* ro1_w = (const float*)d_in[28], *ro1_b = (const float*)d_in[29];
    const float* ro2_w = (const float*)d_in[30], *ro2_b = (const float*)d_in[31];
    const float* ro3_w = (const float*)d_in[32], *ro3_b = (const float*)d_in[33];
    const float* pm1_w = (const float*)d_in[34], *pm1_b = (const float*)d_in[35];
    const float* pm2_w = (const float*)d_in[36], *pm2_b = (const float*)d_in[37];
    const float* pm3_w = (const float*)d_in[38], *pm3_b = (const float*)d_in[39];
    float* out = (float*)d_out;

    const long E = in_sizes[4] / 2;
    const int* src = ei;
    const int* dst = ei + E;

    // ---- workspace bump allocator (16B-aligned chunks) ----
    float* ws = (float*)d_ws;
    size_t off = 0;
    auto allocF = [&](size_t n) { float* p = ws + off; off += (n + 3) & ~(size_t)3; return p; };
    auto allocH = [&](size_t nh) { return (_Float16*)allocF((nh + 1) / 2); };

    // f32 scratch
    float* ALS  = allocF((size_t)NN * HEADS_);
    float* ALD  = allocF((size_t)NN * HEADS_);
    float* MKEYf= allocF((size_t)NN * HEADS_);
    float* ZBUF = allocF((size_t)NN * HEADS_);
    float* EVAL = allocF((size_t)E * HEADS_);
    float* AGG  = allocF((size_t)NN * HD_);
    float* T3   = allocF((size_t)BP * NCLSP);
    unsigned* MKEY = (unsigned*)MKEYf;

    // f16 activations (padded strides / rows)
    _Float16* DRUGN = allocH((size_t)BP * NBITS_);
    _Float16* GEXN  = allocH((size_t)B_ * G_);
    _Float16* DOSEN = allocH(B_);
    _Float16* DURN  = allocH(B_);
    _Float16* DE1   = allocH((size_t)BP * D1P);
    _Float16* DE2   = allocH((size_t)BP * D2P);
    _Float16* DEMB  = allocH((size_t)BP * DRUGP);
    _Float16* XF    = allocH((size_t)NNP * HD_);
    _Float16* H     = allocH((size_t)NNP * HD_);
    _Float16* RT    = allocH((size_t)B_ * CH_ * GP);
    _Float16* O1    = allocH((size_t)B_ * CH_ * R1P);
    _Float16* O2    = allocH((size_t)B_ * CH_ * R2P);
    _Float16* ROUT  = allocH((size_t)B_ * CH_);
    _Float16* TBUF  = allocH((size_t)BP * PREDP);
    _Float16* T1    = allocH((size_t)BP * P1P);
    _Float16* T2    = allocH((size_t)BP * P2P);

    // f16 transposed+padded weights Wt[Np][Kp]
    _Float16* WTdm1 = allocH((size_t)D1P * NBITS_);
    _Float16* WTdm2 = allocH((size_t)D2P * D1P);
    _Float16* WTdm3 = allocH((size_t)DRUGP * D2P);
    _Float16* WTg0  = allocH((size_t)HD_ * DG2VP);
    _Float16* WTg1  = allocH((size_t)HD_ * HD_);
    _Float16* WTro1 = allocH((size_t)R1P * GP);
    _Float16* WTro2 = allocH((size_t)R2P * R1P);
    _Float16* WTpm1 = allocH((size_t)P1P * PREDP);
    _Float16* WTpm2 = allocH((size_t)P2P * P1P);
    _Float16* WTpm3 = allocH((size_t)NCLSP * P2P);

    const int TB = 256;
    auto blocks = [&](long n) { return (unsigned)((n + TB - 1) / TB); };
    auto wconv = [&](const float* W, _Float16* Wt, int K, int N, int Kp, int Np) {
        wconvT<<<blocks((long)Np * Kp), TB, 0, stream>>>(W, Wt, K, N, Kp, Np);
    };

    // 0) weight conversion (f32 [K,N] -> f16 transposed padded [Np,Kp])
    wconv(dm1_w, WTdm1, NBITS_, D1_, NBITS_, D1P);
    wconv(dm2_w, WTdm2, D1_, D2_, D1P, D2P);
    wconv(dm3_w, WTdm3, D2_, DRUG_, D2P, DRUGP);
    wconv(gat0_w, WTg0, DG2V_, HD_, DG2VP, HD_);
    wconv(gat1_w, WTg1, HD_, HD_, HD_, HD_);
    wconv(ro1_w, WTro1, G_, R1_, GP, R1P);
    wconv(ro2_w, WTro2, R1_, R2_, R1P, R2P);
    wconv(pm1_w, WTpm1, PRED_, P1_, PREDP, P1P);
    wconv(pm2_w, WTpm2, P1_, P2_, P1P, P2P);
    wconv(pm3_w, WTpm3, P2_, NCLS_, P2P, NCLSP);

    // 1) batch norms (f16 outputs)
    bn_kernel<<<blocks(NBITS_), TB, 0, stream>>>(drug_in, bn1_g, bn1_b, DRUGN, B_, NBITS_);
    bn_kernel<<<blocks(G_),     TB, 0, stream>>>(gex_in,  bn2_g, bn2_b, GEXN,  B_, G_);
    bn_kernel<<<1, 32, 0, stream>>>(dose, bn3_g, bn3_b, DOSEN, B_, 1);
    bn_kernel<<<1, 32, 0, stream>>>(dur,  bn4_g, bn4_b, DURN,  B_, 1);

    // 2) drug MLP (WMMA, all guard-free)
    launch_gemm_h(DRUGN, NBITS_, WTdm1, NBITS_, dm1_b, D1_, DE1, D1P, BP, D1P, 1, stream);
    launch_gemm_h(DE1, D1P, WTdm2, D1P, dm2_b, D2_, DE2, D2P, BP, D2P, 1, stream);
    launch_gemm_h(DE2, D2P, WTdm3, D2P, dm3_b, DRUG_, DEMB, DRUGP, BP, DRUGP, 1, stream);

    // 3) node features (f16, stride 256, K-pad zeroed)
    gene_feat<<<blocks((long)NN * DG2VP), TB, 0, stream>>>(GEXN, g2v, XF);

    // 4) two GAT hops
    const _Float16* gwt[2] = {WTg0, WTg1};
    const float* gas[2] = {gat0_as, gat1_as};
    const float* gad[2] = {gat0_ad, gat1_ad};
    const float* gb[2]  = {gat0_b, gat1_b};
    int kp[2] = {DG2VP, HD_};
    for (int hop = 0; hop < 2; ++hop) {
        launch_gemm_h(XF, HD_, gwt[hop], kp[hop], nullptr, HD_, H, HD_, NNP, HD_, 0, stream);
        gat_alpha<<<blocks((long)NN * HEADS_), TB, 0, stream>>>(H, gas[hop], gad[hop], ALS, ALD);
        hipMemsetAsync(MKEY, 0, (size_t)NN * HEADS_ * sizeof(unsigned), stream);
        hipMemsetAsync(ZBUF, 0, (size_t)NN * HEADS_ * sizeof(float), stream);
        hipMemsetAsync(AGG,  0, (size_t)NN * HD_ * sizeof(float), stream);
        edge_logits<<<blocks(E * HEADS_), TB, 0, stream>>>(src, dst, ALS, ALD, EVAL, MKEY, E);
        edge_exp   <<<blocks(E * HEADS_), TB, 0, stream>>>(dst, EVAL, MKEY, ZBUF, E);
        edge_agg   <<<blocks(E * HEADS_), TB, 0, stream>>>(src, dst, EVAL, ZBUF, H, AGG, E);
        gat_finish <<<blocks((long)NN * HD_), TB, 0, stream>>>(AGG, gb[hop], XF, RT, hop);
    }

    // 5) readout: rT[16384, 978 (stride 992)] -> 652 -> 326 -> 1
    const int RROWS = B_ * CH_;                         // 16384 (mult of 256)
    launch_gemm_h(RT, GP, WTro1, GP, ro1_b, R1_, O1, R1P, RROWS, R1P, 1, stream);
    launch_gemm_h(O1, R1P, WTro2, R1P, ro2_b, R2_, O2, R2P, RROWS, R2P, 1, stream);
    ro3_dot<<<blocks(RROWS), TB, 0, stream>>>(O2, ro3_w, ro3_b, ROUT, RROWS, R2_, R2P);

    // 6) concat + prediction MLP
    concat_t<<<blocks((long)B_ * PRED_), TB, 0, stream>>>(DEMB, ROUT, DOSEN, DURN, TBUF);
    launch_gemm_h(TBUF, PREDP, WTpm1, PREDP, pm1_b, P1_, T1, P1P, BP, P1P, 1, stream);
    launch_gemm_h(T1, P1P, WTpm2, P1P, pm2_b, P2_, T2, P2P, BP, P2P, 1, stream);
    launch_gemm_f(T2, P2P, WTpm3, P2P, pm3_b, NCLS_, T3, NCLSP, BP, NCLSP, 0, stream);
    copy_out<<<1, 64, 0, stream>>>(T3, out);
}
